// SlotAttention_80882824118351
// MI455X (gfx1250) — matrix operand order repaired and checked by reading, not compiled, not run on hardware
//
#include <hip/hip_runtime.h>

// ---------------------------------------------------------------------------
// Slot Attention for MI455X (gfx1250, wave32, WMMA bf16 16x16x32)
// B=64 N=4096 D=128 K=7 H=256 T=3
// ---------------------------------------------------------------------------

#define BB 64
#define NN 4096
#define DD 128
#define KSLOT 7
#define HH 256
#define TT 3
#define NSPLIT 8
#define SCALE_F 0.08838834764831845f   // 128^-0.5

// gfx1250 async global->LDS path (GLOBAL_LOAD_ASYNC_TO_LDS_B128 + ASYNCcnt).
#define ASYNC_LDS 1

typedef __bf16 bf16_t;
typedef __attribute__((ext_vector_type(16))) __bf16 v16bf;
typedef __attribute__((ext_vector_type(8)))  float  v8f;

// ---------------------------------------------------------------------------
// helpers
// ---------------------------------------------------------------------------
__device__ __forceinline__ float wave_sum(float v) {
#pragma unroll
  for (int m = 16; m >= 1; m >>= 1) v += __shfl_xor(v, m, 32);
  return v;
}

__device__ __forceinline__ v8f wmma_bf16(v16bf a, v16bf b, v8f c) {
  return __builtin_amdgcn_wmma_f32_16x16x32_bf16(
      /*neg_a=*/false, a, /*neg_b=*/false, b,
      /*c_mod=*/(short)0, c, /*reuse_a=*/false, /*reuse_b=*/false);
}

// A fragment, 16x32 bf16 (M x K), row-major source with row stride lda elems.
// lanes 0-15 -> M=lane, K=0..15 ; lanes 16-31 -> M=lane-16, K=16..31  (contiguous)
__device__ __forceinline__ v16bf load_A(const bf16_t* base, int lda, int lane) {
  const int m = lane & 15, half = lane >> 4;
  const bf16_t* row = base + (size_t)m * lda;
  v16bf a;
#pragma unroll
  for (int i = 0; i < 8; ++i) {
    int k0 = half * 16 + i * 2;
    a[2 * i]     = row[k0];
    a[2 * i + 1] = row[k0 + 1];
  }
  return a;
}

// B fragment, 32x16 bf16 (K x N), from a TRANSPOSED source BT[n][k] (stride ldb).
// lanes 0-15 -> N=lane, K=0..15 ; lanes 16-31 -> N=lane-16, K=16..31  (contiguous)
__device__ __forceinline__ v16bf load_BT(const bf16_t* bt, int ldb, int lane, int kbase) {
  const int n = lane & 15, half = lane >> 4;
  const bf16_t* p = bt + (size_t)n * ldb + kbase + half * 16;
  v16bf b;
#pragma unroll
  for (int j = 0; j < 16; ++j) b[j] = p[j];
  return b;
}

// B fragment from a row-major source B[k][n] (stride ldb) - strided gather (LDS)
__device__ __forceinline__ v16bf load_B_rm(const bf16_t* bm, int ldb, int lane,
                                           int kbase, int nbase) {
  const int n = lane & 15, half = lane >> 4;
  const bf16_t* p = bm + (size_t)(kbase + half * 16) * ldb + nbase + n;
  v16bf b;
#pragma unroll
  for (int j = 0; j < 16; ++j) b[j] = p[(size_t)j * ldb];
  return b;
}

// ---------------------------------------------------------------------------
// K0: transpose+convert weights to bf16, copy slots to workspace
// ---------------------------------------------------------------------------
__global__ void k_prep(const float* __restrict__ Wk, const float* __restrict__ Wv,
                       const float* __restrict__ Wq, const float* __restrict__ slots_in,
                       bf16_t* __restrict__ WkT, bf16_t* __restrict__ WvT,
                       bf16_t* __restrict__ WqT, float* __restrict__ slots_ws) {
  int idx = blockIdx.x * blockDim.x + threadIdx.x;
  const int stride = gridDim.x * blockDim.x;
  const int MW = DD * DD;
  for (int i = idx; i < 5 * MW; i += stride) {
    int mat = i / MW, r = i % MW;
    int n = r / DD, k = r % DD;
    const float* src = (mat == 0) ? Wk : (mat == 1) ? Wv : (Wq + (size_t)(mat - 2) * MW);
    bf16_t* dst = (mat == 0) ? WkT : (mat == 1) ? WvT : (WqT + (size_t)(mat - 2) * MW);
    dst[n * DD + k] = (bf16_t)src[k * DD + n];
  }
  for (int i = idx; i < BB * KSLOT * DD; i += stride) slots_ws[i] = slots_in[i];
}

// ---------------------------------------------------------------------------
// K1: fused layernorm(x) -> k = xln@Wk, v = xln@Wv  (bf16 out)
// grid (B, N/128), 256 thr. Wave w owns output row-tile ti=w: its 4 A-frags are
// loaded once and reused for all 16 (2 mat x 8 col) tiles. D staged through LDS
// so the 128 MB k/v write stream goes out as contiguous b128 stores.
// ---------------------------------------------------------------------------
__global__ __launch_bounds__(256) void k_ln_project(
    const float* __restrict__ x, const float* __restrict__ gamma,
    const float* __restrict__ beta, const bf16_t* __restrict__ WkT,
    const bf16_t* __restrict__ WvT, bf16_t* __restrict__ kout,
    bf16_t* __restrict__ vout) {
  __shared__ bf16_t xln[128 * 128];
  __shared__ bf16_t outt[128 * 128];
  const int b = blockIdx.x, nb = blockIdx.y;
  const int tid = threadIdx.x, w = tid >> 5, lane = tid & 31;
  const size_t rowbase = (size_t)b * NN + (size_t)nb * 128;

  // --- layernorm: each wave does 16 rows, lane holds 4 elems/row
  const float4 g  = *(const float4*)(gamma + lane * 4);
  const float4 bt = *(const float4*)(beta + lane * 4);
  for (int rr = 0; rr < 16; ++rr) {
    const int r = w * 16 + rr;
    const float4 xv = *(const float4*)(x + (rowbase + r) * DD + lane * 4);
    float s  = xv.x + xv.y + xv.z + xv.w;
    float s2 = xv.x * xv.x + xv.y * xv.y + xv.z * xv.z + xv.w * xv.w;
    s = wave_sum(s); s2 = wave_sum(s2);
    const float mean = s * (1.0f / DD);
    const float rstd = rsqrtf(s2 * (1.0f / DD) - mean * mean + 1e-6f);
    bf16_t* dst = xln + r * 128 + lane * 4;
    dst[0] = (bf16_t)((xv.x - mean) * rstd * g.x + bt.x);
    dst[1] = (bf16_t)((xv.y - mean) * rstd * g.y + bt.y);
    dst[2] = (bf16_t)((xv.z - mean) * rstd * g.z + bt.z);
    dst[3] = (bf16_t)((xv.w - mean) * rstd * g.w + bt.w);
  }
  __syncthreads();

  const int n = lane & 15, half = lane >> 4;
  // A-fragments for row tile ti = w, reused by all 16 output tiles
  v16bf afrag[4];
#pragma unroll
  for (int kk = 0; kk < 4; ++kk)
    afrag[kk] = load_A(xln + w * 16 * 128 + kk * 32, 128, lane);

  for (int mat = 0; mat < 2; ++mat) {
    const bf16_t* wmatT = mat ? WvT : WkT;
#pragma unroll
    for (int tj = 0; tj < 8; ++tj) {
      const bf16_t* wt = wmatT + (size_t)tj * 16 * DD;
      v8f acc = {0.f, 0.f, 0.f, 0.f, 0.f, 0.f, 0.f, 0.f};
#pragma unroll
      for (int kk = 0; kk < 4; ++kk)
        acc = wmma_bf16(afrag[kk], load_BT(wt, DD, lane, kk * 32), acc);
#pragma unroll
      for (int r = 0; r < 8; ++r)
        outt[(w * 16 + r + 8 * half) * 128 + tj * 16 + n] = (bf16_t)acc[r];
    }
    __syncthreads();
    // coalesced 32 KB block store (this block's rows are contiguous in memory)
    {
      const uint4* s4 = (const uint4*)outt;
      uint4* d4 = (uint4*)((mat ? vout : kout) + rowbase * DD);
#pragma unroll
      for (int i = 0; i < 8; ++i) d4[tid + i * 256] = s4[tid + i * 256];
    }
    __syncthreads();
  }
}

// ---------------------------------------------------------------------------
// K2: s = LN(slots); q = s@Wq[i]; padded to 16 rows, bf16 out. grid B, 256 thr.
// ---------------------------------------------------------------------------
__global__ __launch_bounds__(256) void k_qproj(
    const float* __restrict__ slots_ws, const float* __restrict__ lns,
    const float* __restrict__ lnb, const bf16_t* __restrict__ WqT_i,
    bf16_t* __restrict__ qout) {
  __shared__ bf16_t sln[16 * 128];
  const int b = blockIdx.x, tid = threadIdx.x, w = tid >> 5, lane = tid & 31;
  for (int i = tid; i < 16 * 128; i += 256) sln[i] = (bf16_t)0.0f;
  __syncthreads();
  if (w < KSLOT) {
    const float4 xv = *(const float4*)(slots_ws + ((size_t)b * KSLOT + w) * DD + lane * 4);
    float s  = xv.x + xv.y + xv.z + xv.w;
    float s2 = xv.x * xv.x + xv.y * xv.y + xv.z * xv.z + xv.w * xv.w;
    s = wave_sum(s); s2 = wave_sum(s2);
    const float mean = s * (1.0f / DD);
    const float rstd = rsqrtf(s2 * (1.0f / DD) - mean * mean + 1e-6f);
    const float4 g  = *(const float4*)(lns + lane * 4);
    const float4 bt = *(const float4*)(lnb + lane * 4);
    bf16_t* dst = sln + w * 128 + lane * 4;
    dst[0] = (bf16_t)((xv.x - mean) * rstd * g.x + bt.x);
    dst[1] = (bf16_t)((xv.y - mean) * rstd * g.y + bt.y);
    dst[2] = (bf16_t)((xv.z - mean) * rstd * g.z + bt.z);
    dst[3] = (bf16_t)((xv.w - mean) * rstd * g.w + bt.w);
  }
  __syncthreads();
  v8f acc = {0.f, 0.f, 0.f, 0.f, 0.f, 0.f, 0.f, 0.f};
  const bf16_t* wt = WqT_i + (size_t)w * 16 * DD;
#pragma unroll
  for (int kk = 0; kk < 4; ++kk)
    acc = wmma_bf16(load_A(sln + kk * 32, 128, lane),
                    load_BT(wt, DD, lane, kk * 32), acc);
  const int n = lane & 15, half = lane >> 4;
#pragma unroll
  for (int r = 0; r < 8; ++r)
    qout[((size_t)b * 16 + r + 8 * half) * DD + w * 16 + n] = (bf16_t)acc[r];
}

// ---------------------------------------------------------------------------
// K3: logits = q@k^T * SCALE, softmax over the 7 slots, write attn (f32).
// grid (B, N/256), 256 thr.
// ---------------------------------------------------------------------------
__global__ __launch_bounds__(256) void k_logits_softmax(
    const bf16_t* __restrict__ qmat, const bf16_t* __restrict__ kmat,
    float* __restrict__ attn_out) {
  __shared__ bf16_t qtile[16 * 128];
  __shared__ float lg[16 * 256];
  const int b = blockIdx.x, nb = blockIdx.y;
  const int tid = threadIdx.x, w = tid >> 5, lane = tid & 31;
  {
    const uint4* src = (const uint4*)(qmat + (size_t)b * 16 * DD);
    ((uint4*)qtile)[tid] = src[tid];
  }
  __syncthreads();
  const int n = lane & 15, half = lane >> 4;
#pragma unroll
  for (int tt = 0; tt < 2; ++tt) {
    const int tj = w * 2 + tt;
    const size_t col0 = (size_t)nb * 256 + tj * 16;
    const bf16_t* bbase = kmat + ((size_t)b * NN + col0) * DD;  // BT view of k
    v8f acc = {0.f, 0.f, 0.f, 0.f, 0.f, 0.f, 0.f, 0.f};
#pragma unroll
    for (int kk = 0; kk < 4; ++kk)
      acc = wmma_bf16(load_A(qtile + kk * 32, 128, lane),
                      load_BT(bbase, DD, lane, kk * 32), acc);
#pragma unroll
    for (int r = 0; r < 8; ++r)
      lg[(r + 8 * half) * 256 + tj * 16 + n] = acc[r] * SCALE_F;
  }
  __syncthreads();
  float v[KSLOT];
  float mx = -1e30f;
#pragma unroll
  for (int k = 0; k < KSLOT; ++k) { v[k] = lg[k * 256 + tid]; mx = fmaxf(mx, v[k]); }
  float sum = 0.f;
#pragma unroll
  for (int k = 0; k < KSLOT; ++k) { v[k] = __expf(v[k] - mx); sum += v[k]; }
  const float inv = 1.0f / sum;
  const size_t nglob = (size_t)nb * 256 + tid;
#pragma unroll
  for (int k = 0; k < KSLOT; ++k)
    attn_out[((size_t)b * KSLOT + k) * NN + nglob] = v[k] * inv;
}

// ---------------------------------------------------------------------------
// K4: attn_n = attn / (sum_n attn + eps), bf16, padded to 16 rows.
// grid (B, 16), 256 thr. Deterministic block reduction (no atomics).
// ---------------------------------------------------------------------------
__global__ __launch_bounds__(256) void k_renorm(const float* __restrict__ attn,
                                                bf16_t* __restrict__ attnn) {
  __shared__ float red[8];
  const int b = blockIdx.x, ks = blockIdx.y, tid = threadIdx.x;
  bf16_t* dst = attnn + ((size_t)b * 16 + ks) * NN;
  if (ks >= KSLOT) {
    for (int n = tid; n < NN; n += 256) dst[n] = (bf16_t)0.0f;
    return;
  }
  const float* src = attn + ((size_t)b * KSLOT + ks) * NN;
  float s = 0.f;
  for (int n = tid; n < NN; n += 256) s += src[n];
  s = wave_sum(s);
  if ((tid & 31) == 0) red[tid >> 5] = s;
  __syncthreads();
  if (tid == 0) {
    float t = 0.f;
#pragma unroll
    for (int i = 0; i < 8; ++i) t += red[i];
    red[0] = 1.0f / (t + 1e-8f);
  }
  __syncthreads();
  const float inv = red[0];
  for (int n = tid; n < NN; n += 256) dst[n] = (bf16_t)(src[n] * inv);
}

// ---------------------------------------------------------------------------
// K5: partial updates = attn_n @ v  over an N/NSPLIT stripe.
// grid (B, NSPLIT), 256 thr. v tiles staged to LDS via async-to-LDS b128
// (ASYNCcnt + s_wait_asynccnt), padded stride to break bank conflicts.
// ---------------------------------------------------------------------------
__global__ __launch_bounds__(256) void k_updates(const bf16_t* __restrict__ attnn,
                                                 const bf16_t* __restrict__ vmat,
                                                 float* __restrict__ partial) {
  __shared__ bf16_t vt[32 * 136];
  const int b = blockIdx.x, sp = blockIdx.y;
  const int tid = threadIdx.x, w = tid >> 5, lane = tid & 31;
  const int steps = NN / 32 / NSPLIT;       // 16
  const int kk0 = sp * steps;
  const bf16_t* arow = attnn + (size_t)b * 16 * NN;
  v8f acc = {0.f, 0.f, 0.f, 0.f, 0.f, 0.f, 0.f, 0.f};
  for (int kk = kk0; kk < kk0 + steps; ++kk) {
    const uint4* src = (const uint4*)(vmat + ((size_t)b * NN + (size_t)kk * 32) * DD);
    {
      const int i0 = tid * 2;
#pragma unroll
      for (int j = 0; j < 2; ++j) {
        const int i = i0 + j, row = i >> 4, c = i & 15;  // 16 uint4 per 128-elem row
#if ASYNC_LDS
        unsigned lds_off = (unsigned)(size_t)(vt + row * 136 + c * 8);
        asm volatile("global_load_async_to_lds_b128 %0, %1, off"
                     :: "v"(lds_off), "v"((const void*)(src + i)) : "memory");
#else
        *(uint4*)(vt + row * 136 + c * 8) = src[i];
#endif
      }
#if ASYNC_LDS
      asm volatile("s_wait_asynccnt 0x0" ::: "memory");
#endif
    }
    if (kk + 1 < kk0 + steps)
      __builtin_prefetch(vmat + ((size_t)b * NN + (size_t)(kk + 1) * 32) * DD, 0, 3);
    __syncthreads();
    acc = wmma_bf16(load_A(arow + kk * 32, NN, lane),
                    load_B_rm(vt, 136, lane, 0, w * 16), acc);
    __syncthreads();
  }
  const int n = lane & 15, half = lane >> 4;
#pragma unroll
  for (int r = 0; r < 8; ++r) {
    const int m = r + 8 * half;
    if (m < KSLOT)
      partial[((size_t)sp * BB * KSLOT + (size_t)b * KSLOT + m) * DD + w * 16 + n] = acc[r];
  }
}

__global__ void k_reduce_updates(const float* __restrict__ partial,
                                 float* __restrict__ updates) {
  const int n = BB * KSLOT * DD;
  for (int i = blockIdx.x * blockDim.x + threadIdx.x; i < n;
       i += gridDim.x * blockDim.x) {
    float s = 0.f;
#pragma unroll
    for (int sp = 0; sp < NSPLIT; ++sp) s += partial[(size_t)sp * n + i];
    updates[i] = s;
  }
}

// ---------------------------------------------------------------------------
// K6: GRU update + MLP refine for one (b, slot). grid B*K, 128 thr.
// ---------------------------------------------------------------------------
__global__ __launch_bounds__(128) void k_gru_mlp(
    float* __restrict__ slots_ws, const float* __restrict__ updates,
    const float* __restrict__ Wg, const float* __restrict__ bg,
    const float* __restrict__ Wc, const float* __restrict__ bc,
    const float* __restrict__ mls, const float* __restrict__ mlb,
    const float* __restrict__ W1, const float* __restrict__ b1,
    const float* __restrict__ W2, const float* __restrict__ b2) {
  __shared__ float cat[256], cat2[256], harr[256], hln[128];
  __shared__ float redm[4], redv[4];
  const int blk = blockIdx.x, b = blk / KSLOT, ks = blk % KSLOT;
  const int tid = threadIdx.x;
  float* srow = slots_ws + ((size_t)b * KSLOT + ks) * DD;
  const float* urow = updates + ((size_t)b * KSLOT + ks) * DD;
  const float sprev = srow[tid];
  const float upd_in = urow[tid];
  cat[tid] = upd_in; cat[128 + tid] = sprev;
  __syncthreads();
  float g0 = bg[tid], g1 = bg[128 + tid];
  for (int c = 0; c < 256; ++c) {
    const float cv = cat[c];
    g0 += cv * Wg[(size_t)c * 256 + tid];
    g1 += cv * Wg[(size_t)c * 256 + 128 + tid];
  }
  const float reset = 1.0f / (1.0f + __expf(-g0));
  const float updg  = 1.0f / (1.0f + __expf(-g1));
  cat2[tid] = upd_in; cat2[128 + tid] = reset * sprev;
  __syncthreads();
  float cd = bc[tid];
  for (int c = 0; c < 256; ++c) cd += cat2[c] * Wc[(size_t)c * 128 + tid];
  cd = tanhf(cd);
  const float snew = (1.0f - updg) * sprev + updg * cd;
  float s = wave_sum(snew), s2 = wave_sum(snew * snew);
  if ((tid & 31) == 0) { redm[tid >> 5] = s; redv[tid >> 5] = s2; }
  __syncthreads();
  const float mean = (redm[0] + redm[1] + redm[2] + redm[3]) * (1.0f / 128.0f);
  const float msq  = (redv[0] + redv[1] + redv[2] + redv[3]) * (1.0f / 128.0f);
  const float rstd = rsqrtf(msq - mean * mean + 1e-6f);
  hln[tid] = (snew - mean) * rstd * mls[tid] + mlb[tid];
  __syncthreads();
  float h0 = b1[tid], h1 = b1[128 + tid];
  for (int d = 0; d < 128; ++d) {
    const float hv = hln[d];
    h0 += hv * W1[(size_t)d * 256 + tid];
    h1 += hv * W1[(size_t)d * 256 + 128 + tid];
  }
  harr[tid] = fmaxf(h0, 0.0f);
  harr[128 + tid] = fmaxf(h1, 0.0f);
  __syncthreads();
  float o = b2[tid];
  for (int hh = 0; hh < 256; ++hh) o += harr[hh] * W2[(size_t)hh * 128 + tid];
  srow[tid] = snew + o;
}

__global__ void k_copy(const float* __restrict__ src, float* __restrict__ dst, int n) {
  for (int i = blockIdx.x * blockDim.x + threadIdx.x; i < n; i += gridDim.x * blockDim.x)
    dst[i] = src[i];
}

// ---------------------------------------------------------------------------
extern "C" void kernel_launch(void* const* d_in, const int* in_sizes, int n_in,
                              void* d_out, int out_size, void* d_ws, size_t ws_size,
                              hipStream_t stream) {
  const float* inputs      = (const float*)d_in[0];
  const float* slots       = (const float*)d_in[1];
  const float* ln_in_scale = (const float*)d_in[2];
  const float* ln_in_bias  = (const float*)d_in[3];
  const float* Wk          = (const float*)d_in[4];
  const float* Wv          = (const float*)d_in[5];
  const float* ln_s_scale  = (const float*)d_in[6];
  const float* ln_s_bias   = (const float*)d_in[7];
  const float* Wq          = (const float*)d_in[8];
  const float* Wg          = (const float*)d_in[9];
  const float* bg          = (const float*)d_in[10];
  const float* Wc          = (const float*)d_in[11];
  const float* bc          = (const float*)d_in[12];
  const float* mls         = (const float*)d_in[13];
  const float* mlb         = (const float*)d_in[14];
  const float* W1          = (const float*)d_in[15];
  const float* b1          = (const float*)d_in[16];
  const float* W2          = (const float*)d_in[17];
  const float* b2          = (const float*)d_in[18];
  (void)in_sizes; (void)n_in; (void)out_size; (void)ws_size;

  float* out_slots = (float*)d_out;
  float* out_attn  = out_slots + (size_t)BB * KSLOT * DD;

  char* ws = (char*)d_ws;
  size_t off = 0;
  auto carve = [&](size_t bytes) {
    void* p = ws + off;
    off = (off + bytes + 255) & ~(size_t)255;
    return p;
  };
  bf16_t* WkT      = (bf16_t*)carve((size_t)DD * DD * 2);
  bf16_t* WvT      = (bf16_t*)carve((size_t)DD * DD * 2);
  bf16_t* WqT      = (bf16_t*)carve((size_t)TT * DD * DD * 2);
  float*  slots_ws = (float*) carve((size_t)BB * KSLOT * DD * 4);
  bf16_t* qbf      = (bf16_t*)carve((size_t)BB * 16 * DD * 2);
  bf16_t* attnn    = (bf16_t*)carve((size_t)BB * 16 * NN * 2);
  float*  updates  = (float*) carve((size_t)BB * KSLOT * DD * 4);
  float*  partial  = (float*) carve((size_t)NSPLIT * BB * KSLOT * DD * 4);
  bf16_t* kbf      = (bf16_t*)carve((size_t)BB * NN * DD * 2);
  bf16_t* vbf      = (bf16_t*)carve((size_t)BB * NN * DD * 2);

  k_prep<<<64, 256, 0, stream>>>(Wk, Wv, Wq, slots, WkT, WvT, WqT, slots_ws);
  k_ln_project<<<dim3(BB, NN / 128), 256, 0, stream>>>(
      inputs, ln_in_scale, ln_in_bias, WkT, WvT, kbf, vbf);

  for (int i = 0; i < TT; ++i) {
    k_qproj<<<BB, 256, 0, stream>>>(slots_ws, ln_s_scale + (size_t)i * DD,
                                    ln_s_bias + (size_t)i * DD,
                                    WqT + (size_t)i * DD * DD, qbf);
    k_logits_softmax<<<dim3(BB, NN / 256), 256, 0, stream>>>(qbf, kbf, out_attn);
    k_renorm<<<dim3(BB, 16), 256, 0, stream>>>(out_attn, attnn);
    k_updates<<<dim3(BB, NSPLIT), 256, 0, stream>>>(attnn, vbf, partial);
    k_reduce_updates<<<112, 256, 0, stream>>>(partial, updates);
    k_gru_mlp<<<BB * KSLOT, 128, 0, stream>>>(
        slots_ws, updates,
        Wg + (size_t)i * 256 * 256, bg + (size_t)i * 256,
        Wc + (size_t)i * 256 * 128, bc + (size_t)i * 128,
        mls + (size_t)i * DD, mlb + (size_t)i * DD,
        W1 + (size_t)i * DD * HH, b1 + (size_t)i * HH,
        W2 + (size_t)i * HH * DD, b2 + (size_t)i * DD);
  }
  k_copy<<<64, 256, 0, stream>>>(slots_ws, out_slots, BB * KSLOT * DD);
}
